// AttentionHead_42262478193276
// MI455X (gfx1250) — matrix operand order repaired and checked by reading, not compiled, not run on hardware
//
#include <hip/hip_runtime.h>
#include <hip/hip_bf16.h>

typedef __attribute__((ext_vector_type(16))) __bf16 v16bf;
typedef __attribute__((ext_vector_type(8)))  __bf16 bf16x8;
typedef __attribute__((ext_vector_type(8)))  float  v8f;

#define EDIM 1024
#define HDIM 64
#define BATCH 4
#define SEQ 2048
#define NTOK (BATCH*SEQ)

// ---- fragment loaders -------------------------------------------------------
// A-matrix (16-bit, 16x32), order used here: lane l<16 holds row l, K = 0..15;
// lane l>=16 holds row l-16, K = 16..31.
static __device__ __forceinline__ v16bf load_a_rowmajor_bf16(const __bf16* rowChunk, int half) {
  bf16x8 lo = *(const bf16x8*)(rowChunk + half * 16);
  bf16x8 hi = *(const bf16x8*)(rowChunk + half * 16 + 8);
  v16bf a;
#pragma unroll
  for (int e = 0; e < 8; ++e) { a[e] = lo[e]; a[8 + e] = hi[e]; }
  return a;
}

// B-matrix (16-bit, 32x16), order used here: lane n holds col n, K=0..15; lane n+16 holds col n, K=16..31.
// `p` points at the 16 contiguous elements this lane needs.
static __device__ __forceinline__ v16bf load_b_contig_bf16(const __bf16* p) {
  bf16x8 lo = *(const bf16x8*)(p);
  bf16x8 hi = *(const bf16x8*)(p + 8);
  v16bf b;
#pragma unroll
  for (int e = 0; e < 8; ++e) { b[e] = lo[e]; b[8 + e] = hi[e]; }
  return b;
}

// ---- kernel 0: pack weights once: transpose + f32->bf16 ---------------------
// mats 0..2: Wt[n][k] = W[k*64+n], n in [0,64),  k in [0,1024)   (wq/wk/wv)
// mat  3:    Wt[n][k] = W[k*1024+n], n in [0,1024), k in [0,64)  (wo)
__global__ __launch_bounds__(256)
void pack_weights_kernel(const float* __restrict__ wq, const float* __restrict__ wk,
                         const float* __restrict__ wv, const float* __restrict__ wo,
                         __bf16* __restrict__ Wqt, __bf16* __restrict__ Wkt,
                         __bf16* __restrict__ Wvt, __bf16* __restrict__ Wot)
{
  const int i   = blockIdx.x * 256 + threadIdx.x;   // [0, 65536)
  const int mat = blockIdx.y;
  if (mat < 3) {
    const float* W = (mat == 0) ? wq : (mat == 1) ? wk : wv;
    __bf16* D = (mat == 0) ? Wqt : (mat == 1) ? Wkt : Wvt;
    const int n = i >> 10, k = i & 1023;            // [64][1024]
    D[i] = (__bf16)W[(size_t)k * HDIM + n];
  } else {
    const int n = i >> 6, k = i & 63;               // [1024][64]
    Wot[i] = (__bf16)wo[(size_t)k * EDIM + n];
  }
}

// ---- kernel 1: fused QKV projection, register-blocked + pipelined -----------
// One wave computes a 16-row slab across all 12 output tiles (3 mats x 4 cols).
// Q is pre-scaled by 1/sqrt(H) so the attention kernel skips score scaling.
// Q, K stored row-major bf16 [B*S,64]; V stored transposed per batch [64,S].
__global__ __launch_bounds__(32)
void qkv_proj_kernel(const float* __restrict__ x,
                     const __bf16* __restrict__ Wqt, const __bf16* __restrict__ Wkt,
                     const __bf16* __restrict__ Wvt,
                     const float* __restrict__ bq, const float* __restrict__ bk,
                     const float* __restrict__ bv,
                     __bf16* __restrict__ Qb, __bf16* __restrict__ Kb,
                     __bf16* __restrict__ Vt)
{
  const int l = threadIdx.x, half = l >> 4, ln = l & 15;
  const int r0 = blockIdx.x * 16;                   // flattened token row tile

  v8f acc[12];                                      // [mat*4 + col]
#pragma unroll
  for (int i = 0; i < 12; ++i) acc[i] = {};

  const float* xrow = x + (size_t)(r0 + ln) * EDIM;
  const __bf16* Wt[3] = {Wqt, Wkt, Wvt};
  for (int c = 0; c < EDIM / 32; ++c) {
    // A fragment: convert x row slice f32 -> bf16 (once, reused by 12 WMMAs)
    const float* ap = xrow + c * 32 + half * 16;
    v16bf a;
#pragma unroll
    for (int e = 0; e < 8; ++e) { a[e] = (__bf16)ap[e]; a[8 + e] = (__bf16)ap[8 + e]; }
    // issue ALL 12 B-fragment loads first so waits stagger across the WMMA chain
    v16bf b[12];
#pragma unroll
    for (int mat = 0; mat < 3; ++mat)
#pragma unroll
      for (int col = 0; col < 4; ++col)
        b[mat * 4 + col] = load_b_contig_bf16(Wt[mat] + (size_t)(col * 16 + ln) * EDIM
                                              + c * 32 + half * 16);
#pragma unroll
    for (int i = 0; i < 12; ++i)
      acc[i] = __builtin_amdgcn_wmma_f32_16x16x32_bf16(false, a, false, b[i], (short)0,
                                                       acc[i], false, false);
  }

  const int bidx = r0 / SEQ, s0 = r0 % SEQ;         // tile never crosses batches
#pragma unroll
  for (int col = 0; col < 4; ++col) {
    const float bbq = bq[col * 16 + ln];
    const float bbk = bk[col * 16 + ln];
    const float bbv = bv[col * 16 + ln];
#pragma unroll
    for (int r = 0; r < 8; ++r) {
      const int m = r + half * 8;                   // C layout: reg r -> rows r / r+8
      Qb[(size_t)(r0 + m) * HDIM + col * 16 + ln] =
          (__bf16)((acc[0 * 4 + col][r] + bbq) * 0.125f);   // fold 1/sqrt(64)
      Kb[(size_t)(r0 + m) * HDIM + col * 16 + ln] = (__bf16)(acc[1 * 4 + col][r] + bbk);
      Vt[((size_t)bidx * HDIM + col * 16 + ln) * SEQ + s0 + m] =
          (__bf16)(acc[2 * 4 + col][r] + bbv);
    }
  }
}

// ---- kernel 2: causal flash attention, one wave per 16-row query tile -------
__global__ __launch_bounds__(32)
void flash_attn_kernel(const __bf16* __restrict__ Qb, const __bf16* __restrict__ Kb,
                       const __bf16* __restrict__ Vt, __bf16* __restrict__ Zb)
{
  const int l = threadIdx.x, half = l >> 4, ln = l & 15;
  const int qt   = blockIdx.x;     // query tile [0, 128)
  const int bidx = blockIdx.y;     // batch
  __shared__ __bf16 Plds[16 * 32]; // P tile staged C-layout -> A-layout

  const size_t qrow = (size_t)bidx * SEQ + (size_t)qt * 16;
  v16bf aq[2];                     // Q pre-scaled by 1/sqrt(H) in kernel 1
  aq[0] = load_a_rowmajor_bf16(Qb + (qrow + ln) * HDIM + 0, half);
  aq[1] = load_a_rowmajor_bf16(Qb + (qrow + ln) * HDIM + 32, half);

  float mrow[8], lsum[8];
  v8f o0 = {}, o1 = {}, o2 = {}, o3 = {};
#pragma unroll
  for (int r = 0; r < 8; ++r) { mrow[r] = -1e30f; lsum[r] = 0.0f; }

  const int nblk = (qt >> 1) + 1;            // 32-key blocks covering keys <= qmax
  for (int j2 = 0; j2 < nblk; ++j2) {
    const int key0 = j2 * 32;

    // prefetch next block's K rows / V rows into cache (global_prefetch_b8)
    if (j2 + 1 < nblk) {
      __builtin_prefetch(Kb + ((size_t)bidx * SEQ + key0 + 32 + l) * HDIM, 0, 3);
      __builtin_prefetch(Vt + ((size_t)bidx * HDIM + l * 2) * SEQ + key0 + 32, 0, 3);
    }

    // ---- scores: two 16x16 tiles over this 32-key block ----
    v8f slo = {}, shi = {};
#pragma unroll
    for (int c = 0; c < 2; ++c) {            // contraction over H in K=32 chunks
      const __bf16* kbase = Kb + ((size_t)bidx * SEQ + key0) * HDIM + c * 32 + half * 16;
      v16bf bklo = load_b_contig_bf16(kbase + (size_t)ln * HDIM);
      v16bf bkhi = load_b_contig_bf16(kbase + (size_t)(16 + ln) * HDIM);
      slo = __builtin_amdgcn_wmma_f32_16x16x32_bf16(false, aq[c], false, bklo, (short)0, slo, false, false);
      shi = __builtin_amdgcn_wmma_f32_16x16x32_bf16(false, aq[c], false, bkhi, (short)0, shi, false, false);
    }

    // issue V B-fragment loads NOW; softmax VALU below hides their latency
    const __bf16* vbase = Vt + (size_t)bidx * HDIM * SEQ + key0 + half * 16;
    v16bf bv0 = load_b_contig_bf16(vbase + (size_t)(0 * 16 + ln) * SEQ);
    v16bf bv1 = load_b_contig_bf16(vbase + (size_t)(1 * 16 + ln) * SEQ);
    v16bf bv2 = load_b_contig_bf16(vbase + (size_t)(2 * 16 + ln) * SEQ);
    v16bf bv3 = load_b_contig_bf16(vbase + (size_t)(3 * 16 + ln) * SEQ);

    // ---- causal mask + online softmax (rows spread across 16 lanes/half) ----
    float alpha[8];
#pragma unroll
    for (int r = 0; r < 8; ++r) {
      const int q = qt * 16 + r + half * 8;
      float vlo = slo[r], vhi = shi[r];
      if (key0 + ln      > q) vlo = -1e30f;
      if (key0 + 16 + ln > q) vhi = -1e30f;
      float t = fmaxf(vlo, vhi);
      t = fmaxf(t, __shfl_xor(t, 1)); t = fmaxf(t, __shfl_xor(t, 2));
      t = fmaxf(t, __shfl_xor(t, 4)); t = fmaxf(t, __shfl_xor(t, 8));
      const float mnew = fmaxf(mrow[r], t);
      const float a   = __expf(mrow[r] - mnew);
      const float plo = __expf(vlo - mnew);
      const float phi = __expf(vhi - mnew);
      float ps = plo + phi;
      ps += __shfl_xor(ps, 1); ps += __shfl_xor(ps, 2);
      ps += __shfl_xor(ps, 4); ps += __shfl_xor(ps, 8);
      lsum[r] = lsum[r] * a + ps;
      mrow[r] = mnew; alpha[r] = a;
      const int m = r + half * 8;
      Plds[m * 32 + ln]      = (__bf16)plo;  // P[m][k],   k = key offset 0..15
      Plds[m * 32 + 16 + ln] = (__bf16)phi;  // P[m][k+16]
    }
#pragma unroll
    for (int r = 0; r < 8; ++r) {
      o0[r] *= alpha[r]; o1[r] *= alpha[r]; o2[r] *= alpha[r]; o3[r] *= alpha[r];
    }
    __syncthreads();                         // single-wave WG: ordering fence only

    // ---- P (16x32 bf16 A) @ V (32x64) via 4 WMMAs ----
    v16bf pa = load_a_rowmajor_bf16(Plds + ln * 32, half);
    o0 = __builtin_amdgcn_wmma_f32_16x16x32_bf16(false, pa, false, bv0, (short)0, o0, false, false);
    o1 = __builtin_amdgcn_wmma_f32_16x16x32_bf16(false, pa, false, bv1, (short)0, o1, false, false);
    o2 = __builtin_amdgcn_wmma_f32_16x16x32_bf16(false, pa, false, bv2, (short)0, o2, false, false);
    o3 = __builtin_amdgcn_wmma_f32_16x16x32_bf16(false, pa, false, bv3, (short)0, o3, false, false);
    __syncthreads();
  }

  // ---- normalize and store Z (bf16, row-major [B*S,64]) ----
#pragma unroll
  for (int r = 0; r < 8; ++r) {
    const float inv = 1.0f / lsum[r];
    const int m = r + half * 8;
    __bf16* zr = Zb + (qrow + m) * HDIM;
    zr[0 * 16 + ln] = (__bf16)(o0[r] * inv);
    zr[1 * 16 + ln] = (__bf16)(o1[r] * inv);
    zr[2 * 16 + ln] = (__bf16)(o2[r] * inv);
    zr[3 * 16 + ln] = (__bf16)(o3[r] * inv);
  }
}

// ---- kernel 3: output projection Z @ wo + bo (f32 out), 16x64 slab per wave -
__global__ __launch_bounds__(32)
void out_proj_kernel(const __bf16* __restrict__ Zb, const __bf16* __restrict__ Wot,
                     const float* __restrict__ bo, float* __restrict__ out)
{
  const int l = threadIdx.x, half = l >> 4, ln = l & 15;
  const int r0   = blockIdx.x * 16;
  const int col0 = blockIdx.y * 64;

  // preload both A fragments and all 8 B fragments, then run the WMMA chain
  v16bf a0 = load_a_rowmajor_bf16(Zb + (size_t)(r0 + ln) * HDIM + 0, half);
  v16bf a1 = load_a_rowmajor_bf16(Zb + (size_t)(r0 + ln) * HDIM + 32, half);
  v16bf b[8];
#pragma unroll
  for (int c = 0; c < 2; ++c)
#pragma unroll
    for (int t = 0; t < 4; ++t)
      b[c * 4 + t] = load_b_contig_bf16(Wot + (size_t)(col0 + t * 16 + ln) * HDIM
                                        + c * 32 + half * 16);
  v8f acc[4];
#pragma unroll
  for (int t = 0; t < 4; ++t) acc[t] = {};
#pragma unroll
  for (int c = 0; c < 2; ++c)
#pragma unroll
    for (int t = 0; t < 4; ++t)
      acc[t] = __builtin_amdgcn_wmma_f32_16x16x32_bf16(false, (c ? a1 : a0), false,
                                                       b[c * 4 + t], (short)0,
                                                       acc[t], false, false);
#pragma unroll
  for (int t = 0; t < 4; ++t) {
    const float bb = bo[col0 + t * 16 + ln];
#pragma unroll
    for (int r = 0; r < 8; ++r)
      out[(size_t)(r0 + r + half * 8) * EDIM + col0 + t * 16 + ln] = acc[t][r] + bb;
  }
}

// ---- host launch ------------------------------------------------------------
extern "C" void kernel_launch(void* const* d_in, const int* in_sizes, int n_in,
                              void* d_out, int out_size, void* d_ws, size_t ws_size,
                              hipStream_t stream) {
  const float* x  = (const float*)d_in[0];
  const float* wq = (const float*)d_in[1];
  const float* bq = (const float*)d_in[2];
  const float* wk = (const float*)d_in[3];
  const float* bk = (const float*)d_in[4];
  const float* wv = (const float*)d_in[5];
  const float* bv = (const float*)d_in[6];
  const float* wo = (const float*)d_in[7];
  const float* bo = (const float*)d_in[8];
  float* out = (float*)d_out;

  char* wsc = (char*)d_ws;                          // 4.5 MB used
  const size_t TOKB = (size_t)NTOK * HDIM * 2;      // 1 MB
  __bf16* Qb  = (__bf16*)(wsc + 0 * TOKB);
  __bf16* Kb  = (__bf16*)(wsc + 1 * TOKB);
  __bf16* Vt  = (__bf16*)(wsc + 2 * TOKB);
  __bf16* Zb  = (__bf16*)(wsc + 3 * TOKB);
  __bf16* Wqt = (__bf16*)(wsc + 4 * TOKB);          // each 128 KB
  __bf16* Wkt = Wqt + (size_t)EDIM * HDIM;
  __bf16* Wvt = Wkt + (size_t)EDIM * HDIM;
  __bf16* Wot = Wvt + (size_t)EDIM * HDIM;

  pack_weights_kernel<<<dim3(EDIM * HDIM / 256, 4), 256, 0, stream>>>(
      wq, wk, wv, wo, Wqt, Wkt, Wvt, Wot);
  qkv_proj_kernel<<<dim3(NTOK / 16), 32, 0, stream>>>(
      x, Wqt, Wkt, Wvt, bq, bk, bv, Qb, Kb, Vt);
  flash_attn_kernel<<<dim3(SEQ / 16, BATCH), 32, 0, stream>>>(Qb, Kb, Vt, Zb);
  out_proj_kernel<<<dim3(NTOK / 16, EDIM / 64), 32, 0, stream>>>(Zb, Wot, bo, out);
}
